// HungarianMatcher_48232482734482
// MI455X (gfx1250) — matrix operand order repaired and checked by reading, not compiled, not run on hardware
//
#include <hip/hip_runtime.h>
#include <hip/hip_bf16.h>

// ---- problem constants (from reference setup_inputs) ----
constexpr int BB   = 2;
constexpr int NQ   = 100;
constexpr int CC   = 134;
constexpr int NGT  = 50;
constexpr int HW   = 256 * 256;
constexpr int PP   = 12544;           // multiple of 32 (392 * 32), and of 128
constexpr int NQ_PAD = 112;           // 7 * 16
constexpr int NG_PAD = 64;            // 4 * 16

constexpr float K_ALPHA   = 0.25f;
constexpr float K_CCLASS  = 2.0f;
constexpr float K_CMASK   = 5.0f;
constexpr float K_CDICE   = 5.0f;
constexpr float K_EPS     = 1e-4f;

typedef __attribute__((ext_vector_type(8)))  _Float16 v8h;
typedef __attribute__((ext_vector_type(16))) _Float16 v16h;
typedef __attribute__((ext_vector_type(8)))  float    v8f;

static __device__ __forceinline__ v16h cat16(v8h lo, v8h hi) {
  return __builtin_shufflevector(lo, hi, 0,1,2,3,4,5,6,7,8,9,10,11,12,13,14,15);
}

static __device__ __forceinline__ float softplus_f(float x) {
  // log(1 + exp(x)), numerically stable
  float r = log1pf(__expf(-fabsf(x)));
  return x > 0.0f ? x + r : r;
}

// ---------------------------------------------------------------------------
// Kernel 1: gather predicted mask logits at sampled points, compute focal
// terms; write f16 GEMM operands + per-row reductions. Zero-fills pad rows.
// grid (NQ_PAD, B), block 256
// ---------------------------------------------------------------------------
__global__ void hm_gather_pred(const float* __restrict__ mask_logits,
                               const int* __restrict__ point_idx,
                               _Float16* __restrict__ Adiff,   // [B][NQ_PAD][P]
                               _Float16* __restrict__ Ap,      // [B][NQ_PAD][P]
                               float* __restrict__ SumNeg,     // [B][NQ]
                               float* __restrict__ Psum) {     // [B][NQ]
  const int q = blockIdx.x;
  const int b = blockIdx.y;
  const int tid = threadIdx.x;
  _Float16* ad = Adiff + ((size_t)b * NQ_PAD + q) * PP;
  _Float16* ap = Ap    + ((size_t)b * NQ_PAD + q) * PP;

  if (q >= NQ) {  // pad rows -> zeros so GEMM edge tiles are exact
    for (int p = tid; p < PP; p += blockDim.x) { ad[p] = (_Float16)0.f; ap[p] = (_Float16)0.f; }
    return;
  }

  const float* row = mask_logits + ((size_t)b * NQ + q) * HW;
  const int* idx = point_idx + (size_t)b * PP;

  float sneg = 0.0f, sp = 0.0f;
  for (int p = tid; p < PP; p += blockDim.x) {
    float om  = row[idx[p]];
    float pr  = 1.0f / (1.0f + __expf(-om));     // sigmoid
    float cep = softplus_f(-om);                 // -log(p)
    float cen = softplus_f(om);                  // -log(1-p)
    float omp = 1.0f - pr;
    float fpos = K_ALPHA * omp * omp * cep;
    float fneg = (1.0f - K_ALPHA) * pr * pr * cen;
    ad[p] = (_Float16)(fpos - fneg);
    ap[p] = (_Float16)pr;
    sneg += fneg;
    sp   += pr;
  }

  __shared__ float s1[256], s2[256];
  s1[tid] = sneg; s2[tid] = sp;
  __syncthreads();
  for (int off = 128; off > 0; off >>= 1) {
    if (tid < off) { s1[tid] += s1[tid + off]; s2[tid] += s2[tid + off]; }
    __syncthreads();
  }
  if (tid == 0) {
    SumNeg[b * NQ + q] = s1[0];
    Psum[b * NQ + q]   = s2[0];
  }
}

// ---------------------------------------------------------------------------
// Kernel 2: gather target masks at sampled points -> f16 + row sums.
// grid (NG_PAD, B), block 256
// ---------------------------------------------------------------------------
__global__ void hm_gather_tgt(const int* __restrict__ tgt_masks,
                              const int* __restrict__ point_idx,
                              _Float16* __restrict__ Btm,   // [B][NG_PAD][P]
                              float* __restrict__ Tsum) {   // [B][NGT]
  const int g = blockIdx.x;
  const int b = blockIdx.y;
  const int tid = threadIdx.x;
  _Float16* bt = Btm + ((size_t)b * NG_PAD + g) * PP;

  if (g >= NGT) {
    for (int p = tid; p < PP; p += blockDim.x) bt[p] = (_Float16)0.f;
    return;
  }

  const int* row = tgt_masks + ((size_t)b * NGT + g) * HW;
  const int* idx = point_idx + (size_t)b * PP;

  float s = 0.0f;
  for (int p = tid; p < PP; p += blockDim.x) {
    float t = (float)row[idx[p]];   // {0,1}
    bt[p] = (_Float16)t;
    s += t;
  }

  __shared__ float s1[256];
  s1[tid] = s;
  __syncthreads();
  for (int off = 128; off > 0; off >>= 1) {
    if (tid < off) s1[tid] += s1[tid + off];
    __syncthreads();
  }
  if (tid == 0) Tsum[b * NGT + g] = s1[0];
}

// ---------------------------------------------------------------------------
// Kernel 3: classification cost = -2 * softmax(logits)[label_g]
// grid (NQ, B), block 256
// ---------------------------------------------------------------------------
__global__ void hm_class_cost(const float* __restrict__ class_logits,
                              const int* __restrict__ tgt_labels,
                              float* __restrict__ ccls) {   // [B][NQ][NGT]
  const int q = blockIdx.x;
  const int b = blockIdx.y;
  const int tid = threadIdx.x;

  __shared__ float sl[CC];
  __shared__ float red[256];

  const float* lg = class_logits + ((size_t)b * NQ + q) * CC;
  if (tid < CC) sl[tid] = lg[tid];
  __syncthreads();

  float m = -3.4e38f;
  for (int c = tid; c < CC; c += blockDim.x) m = fmaxf(m, sl[c]);
  red[tid] = m;
  __syncthreads();
  for (int off = 128; off > 0; off >>= 1) {
    if (tid < off) red[tid] = fmaxf(red[tid], red[tid + off]);
    __syncthreads();
  }
  const float mx = red[0];
  __syncthreads();

  float s = 0.0f;
  for (int c = tid; c < CC; c += blockDim.x) s += __expf(sl[c] - mx);
  red[tid] = s;
  __syncthreads();
  for (int off = 128; off > 0; off >>= 1) {
    if (tid < off) red[tid] += red[tid + off];
    __syncthreads();
  }
  const float denom = red[0];
  __syncthreads();

  if (tid < NGT) {
    int lab = tgt_labels[b * NGT + tid];
    float prob = __expf(sl[lab] - mx) / denom;
    ccls[((size_t)b * NQ + q) * NGT + tid] = -K_CCLASS * prob;
  }
}

// ---------------------------------------------------------------------------
// Kernel 4: dual WMMA GEMM over K=P with fused epilogue.
//   C1[q][g] = (focal_pos - focal_neg) . tm   C2[q][g] = p . tm
//   out = ccls + CMASK/P*(SumNeg + C1) + CDICE*(1 - (2*C2+eps)/(Psum+Tsum+eps))
// grid (NG_PAD/16, NQ_PAD/16, B), block 32 (one wave)
// VGPR layouts per CDNA5 ISA 7.12.2:
//   A (16x32 f16): lane l<16 row M=l holds K {k0..k0+7, k0+16..k0+23};
//                  lane l>=16 row M=l-16 holds K {k0+8..15, k0+24..31}
//   B (32x16 f16): lane l holds column N=l%16, same K split as A:
//                  {k0+h8..k0+h8+7, k0+h8+16..k0+h8+23}, h8 = (l<16 ? 0 : 8)
//   C/D (f32):     lane l: N=l%16; VGPR r: M = r + (l<16 ? 0 : 8)
// ---------------------------------------------------------------------------
__global__ __launch_bounds__(32)
void hm_wmma_gemm(const _Float16* __restrict__ Adiff,
                  const _Float16* __restrict__ Ap,
                  const _Float16* __restrict__ Btm,
                  const float* __restrict__ SumNeg,
                  const float* __restrict__ Psum,
                  const float* __restrict__ Tsum,
                  const float* __restrict__ ccls,
                  float* __restrict__ out) {
  const int gt = blockIdx.x;      // N tile
  const int qt = blockIdx.y;      // M tile
  const int b  = blockIdx.z;
  const int lane = threadIdx.x;   // 0..31
  const int half = lane >> 4;     // 0 or 1
  const int l16  = lane & 15;

  const _Float16* a1row = Adiff + ((size_t)b * NQ_PAD + qt * 16 + l16) * PP;
  const _Float16* a2row = Ap    + ((size_t)b * NQ_PAD + qt * 16 + l16) * PP;
  const _Float16* brow  = Btm   + ((size_t)b * NG_PAD + gt * 16 + l16) * PP;

  const int aoff = half * 8;    // per-lane K sub-offset for A
  const int boff = half * 8;    // per-lane K sub-offset for B

  v8f acc1 = {};
  v8f acc2 = {};

#pragma unroll 4
  for (int k0 = 0; k0 < PP; k0 += 32) {
    v8h a1lo = *(const v8h*)(a1row + k0 + aoff);
    v8h a1hi = *(const v8h*)(a1row + k0 + aoff + 16);
    v8h a2lo = *(const v8h*)(a2row + k0 + aoff);
    v8h a2hi = *(const v8h*)(a2row + k0 + aoff + 16);
    v16h bm  = cat16(*(const v8h*)(brow + k0 + boff), *(const v8h*)(brow + k0 + boff + 16));
    v16h a1  = cat16(a1lo, a1hi);
    v16h a2  = cat16(a2lo, a2hi);
    acc1 = __builtin_amdgcn_wmma_f32_16x16x32_f16(false, a1, false, bm,
                                                  (short)0, acc1, false, false);
    acc2 = __builtin_amdgcn_wmma_f32_16x16x32_f16(false, a2, false, bm,
                                                  (short)0, acc2, false, false);
  }

  const int g = gt * 16 + l16;
  if (g >= NGT) return;
  const float tsum = Tsum[b * NGT + g];
  const float inv_p = K_CMASK / (float)PP;

#pragma unroll
  for (int r = 0; r < 8; ++r) {
    int q = qt * 16 + r + half * 8;
    if (q >= NQ) continue;
    size_t oidx = ((size_t)b * NQ + q) * NGT + g;
    float c1 = acc1[r];
    float c2 = acc2[r];
    float maskc = inv_p * (SumNeg[b * NQ + q] + c1);
    float dice  = K_CDICE * (1.0f - (2.0f * c2 + K_EPS) /
                                    (Psum[b * NQ + q] + tsum + K_EPS));
    out[oidx] = ccls[oidx] + maskc + dice;
  }
}

// ---------------------------------------------------------------------------
static inline size_t align256(size_t x) { return (x + 255) & ~(size_t)255; }

extern "C" void kernel_launch(void* const* d_in, const int* in_sizes, int n_in,
                              void* d_out, int out_size, void* d_ws, size_t ws_size,
                              hipStream_t stream) {
  const float* class_logits = (const float*)d_in[0];   // [B,NQ,C]
  const float* mask_logits  = (const float*)d_in[1];   // [B,NQ,H,W]
  const int*   tgt_labels   = (const int*)d_in[2];     // [B,NGT]
  const int*   tgt_masks    = (const int*)d_in[3];     // [B,NGT,H,W]
  const int*   point_idx    = (const int*)d_in[4];     // [B,P]
  float* out = (float*)d_out;                          // [B,NQ,NGT]

  // workspace carve-up
  char* w = (char*)d_ws;
  size_t off = 0;
  _Float16* Adiff = (_Float16*)(w + off); off += align256((size_t)BB * NQ_PAD * PP * sizeof(_Float16));
  _Float16* Ap    = (_Float16*)(w + off); off += align256((size_t)BB * NQ_PAD * PP * sizeof(_Float16));
  _Float16* Btm   = (_Float16*)(w + off); off += align256((size_t)BB * NG_PAD * PP * sizeof(_Float16));
  float* SumNeg   = (float*)(w + off);    off += align256((size_t)BB * NQ * sizeof(float));
  float* Psum     = (float*)(w + off);    off += align256((size_t)BB * NQ * sizeof(float));
  float* Tsum     = (float*)(w + off);    off += align256((size_t)BB * NGT * sizeof(float));
  float* Ccls     = (float*)(w + off);    off += align256((size_t)BB * NQ * NGT * sizeof(float));
  (void)ws_size; (void)n_in; (void)in_sizes; (void)out_size;

  hm_gather_pred<<<dim3(NQ_PAD, BB), 256, 0, stream>>>(
      mask_logits, point_idx, Adiff, Ap, SumNeg, Psum);

  hm_gather_tgt<<<dim3(NG_PAD, BB), 256, 0, stream>>>(
      tgt_masks, point_idx, Btm, Tsum);

  hm_class_cost<<<dim3(NQ, BB), 256, 0, stream>>>(
      class_logits, tgt_labels, Ccls);

  hm_wmma_gemm<<<dim3(NG_PAD / 16, NQ_PAD / 16, BB), 32, 0, stream>>>(
      Adiff, Ap, Btm, SumNeg, Psum, Tsum, Ccls, out);
}
